// GCRN_85066122265299
// MI455X (gfx1250) — matrix operand order repaired and checked by reading, not compiled, not run on hardware
//
#include <hip/hip_runtime.h>
#include <hip/hip_bf16.h>
#include <math.h>

// ---------------------------------------------------------------------------
// Problem constants (from reference): B=32, T=24, N=1024, DIN=2, H=64, E=10, K=3
// ---------------------------------------------------------------------------
#define BB   32
#define TT   24
#define NN   1024
#define DIN  2
#define HH   64
#define EE   10
#define KK   3

// Layer 0: Cin = DIN+H = 66 -> pad 96 ; KC = 3*96  = 288
// Layer 1: Cin = 2H    = 128          ; KC = 3*128 = 384
#define CPAD0 96
#define CPAD1 128
#define KC0   (KK*CPAD0)
#define KC1   (KK*CPAD1)

typedef __attribute__((ext_vector_type(16))) __bf16 v16bf;
typedef __attribute__((ext_vector_type(8)))  __bf16 v8bf;
typedef __attribute__((ext_vector_type(8)))  float  v8f;

__device__ __forceinline__ v16bf cat16(v8bf lo, v8bf hi) {
    return __builtin_shufflevector(lo, hi, 0,1,2,3,4,5,6,7,8,9,10,11,12,13,14,15);
}
__device__ __forceinline__ v8f wmma_bf16(v16bf a, v16bf b, v8f c) {
    return __builtin_amdgcn_wmma_f32_16x16x32_bf16(false, a, false, b, (short)0, c,
                                                   false, false);
}
__device__ __forceinline__ float sigmoidf_(float x) { return 1.f / (1.f + __expf(-x)); }

// ---------------------------------------------------------------------------
// Adjacency: A[n, :] = softmax_j( relu(E[n] . E[j]) )   -> bf16 [N,N]
// ---------------------------------------------------------------------------
__global__ void adjacency_kernel(const float* __restrict__ Emb, __bf16* __restrict__ A) {
    __shared__ float red[256];
    __shared__ float en[EE];
    const int n = blockIdx.x, tid = threadIdx.x;
    if (tid < EE) en[tid] = Emb[n * EE + tid];
    __syncthreads();

    float l[4];
    float mx = 0.f;                       // relu >= 0, so row max >= 0
    #pragma unroll
    for (int q = 0; q < 4; ++q) {
        const int j = tid + q * 256;
        float d = 0.f;
        #pragma unroll
        for (int e = 0; e < EE; ++e) d += en[e] * Emb[j * EE + e];
        d = fmaxf(d, 0.f);
        l[q] = d;
        mx = fmaxf(mx, d);
    }
    red[tid] = mx; __syncthreads();
    for (int s = 128; s > 0; s >>= 1) {
        if (tid < s) red[tid] = fmaxf(red[tid], red[tid + s]);
        __syncthreads();
    }
    const float rowmax = red[0]; __syncthreads();

    float ex[4], sum = 0.f;
    #pragma unroll
    for (int q = 0; q < 4; ++q) { ex[q] = __expf(l[q] - rowmax); sum += ex[q]; }
    red[tid] = sum; __syncthreads();
    for (int s = 128; s > 0; s >>= 1) {
        if (tid < s) red[tid] += red[tid + s];
        __syncthreads();
    }
    const float inv = 1.f / red[0];
    #pragma unroll
    for (int q = 0; q < 4; ++q)
        A[(size_t)n * NN + tid + q * 256] = (__bf16)(ex[q] * inv);
}

// ---------------------------------------------------------------------------
// bf16 tile transpose through LDS: src[batch][R][C] -> dst[batch][C][R]
// ---------------------------------------------------------------------------
__global__ void transpose_bf16(const __bf16* __restrict__ src, __bf16* __restrict__ dst,
                               int R, int C) {
    __shared__ unsigned short tile[32][33];
    const size_t base = (size_t)blockIdx.z * R * C;
    const __bf16* s = src + base;
    __bf16*       d = dst + base;
    const int c0 = blockIdx.x * 32, r0 = blockIdx.y * 32;
    const int tx = threadIdx.x, ty = threadIdx.y;     // (32,8)
    for (int j = ty; j < 32; j += 8) {
        const int r = r0 + j, c = c0 + tx;
        tile[j][tx] = (r < R && c < C) ? *(const unsigned short*)&s[(size_t)r * C + c]
                                       : (unsigned short)0;
    }
    __syncthreads();
    for (int j = ty; j < 32; j += 8) {
        const int c = c0 + j, r = r0 + tx;
        if (c < C && r < R)
            *(unsigned short*)&d[(size_t)c * R + r] = tile[tx][j];
    }
}

// ---------------------------------------------------------------------------
// T2 = 2*A@A - I  (WMMA bf16, f32 accumulate).  grid (N/16, N/64), block 32.
// ---------------------------------------------------------------------------
__global__ void cheb2_kernel(const __bf16* __restrict__ A, const __bf16* __restrict__ AT,
                             __bf16* __restrict__ T2) {
    const int lane = threadIdx.x;
    const int mt = blockIdx.x, cg = blockIdx.y;
    const int row  = mt * 16 + (lane & 15);
    const int aoff = (lane >> 4) * 8;
    const int boff = (lane >> 4) * 8;
    v8f acc[4];
    #pragma unroll
    for (int i = 0; i < 4; ++i) acc[i] = (v8f){0.f,0.f,0.f,0.f,0.f,0.f,0.f,0.f};

    for (int kk = 0; kk < NN; kk += 32) {
        const __bf16* ap = A + (size_t)row * NN + kk + aoff;
        const v16bf af = cat16(*(const v8bf*)ap, *(const v8bf*)(ap + 16));
        #pragma unroll
        for (int ct = 0; ct < 4; ++ct) {
            const int col = cg * 64 + ct * 16 + (lane & 15);
            const __bf16* bp = AT + (size_t)col * NN + kk + boff;
            const v16bf bf = cat16(*(const v8bf*)bp, *(const v8bf*)(bp + 16));
            acc[ct] = wmma_bf16(af, bf, acc[ct]);
        }
    }
    #pragma unroll
    for (int ct = 0; ct < 4; ++ct)
        #pragma unroll
        for (int i = 0; i < 8; ++i) {
            const int m   = mt * 16 + i + ((lane >> 4) ? 8 : 0);
            const int col = cg * 64 + ct * 16 + (lane & 15);
            const float v = 2.f * acc[ct][i] - (m == col ? 1.f : 0.f);
            T2[(size_t)m * NN + col] = (__bf16)v;
        }
}

// ---------------------------------------------------------------------------
// Per-node weights, pre-transposed: Wt[n][o][k*Cpad+c] = sum_d E[n,d]*w[d,k,c,o]
// ---------------------------------------------------------------------------
__global__ void mk_weights(const float* __restrict__ Emb, const float* __restrict__ w,
                           __bf16* __restrict__ Wt, int Cin, int Cpad, int O, int total) {
    const int KCp = KK * Cpad;
    for (int idx = blockIdx.x * blockDim.x + threadIdx.x; idx < total;
         idx += gridDim.x * blockDim.x) {
        const int kc = idx % KCp;
        const int o  = (idx / KCp) % O;
        const int n  = idx / (KCp * O);
        const int k  = kc / Cpad, c = kc % Cpad;
        float v = 0.f;
        if (c < Cin) {
            #pragma unroll
            for (int d = 0; d < EE; ++d)
                v += Emb[n * EE + d] * w[(((size_t)d * KK + k) * Cin + c) * O + o];
        }
        Wt[idx] = (__bf16)v;
    }
}

__global__ void mk_bias(const float* __restrict__ Emb, const float* __restrict__ bsrc,
                        float* __restrict__ dst, int O) {
    const int idx = blockIdx.x * blockDim.x + threadIdx.x;
    if (idx >= NN * O) return;
    const int o = idx % O, n = idx / O;
    float v = 0.f;
    #pragma unroll
    for (int d = 0; d < EE; ++d) v += Emb[n * EE + d] * bsrc[d * O + o];
    dst[idx] = v;
}

// ---------------------------------------------------------------------------
// Pack [x_t | h]  (stage 0)  or  [x_t | z*h] (stage 1) -> bf16, channel-padded;
// also fills the k=0 (identity support) slot of xg.
// ---------------------------------------------------------------------------
__global__ void pack_kernel(const float* __restrict__ xf, const __bf16* __restrict__ ybf,
                            const float* __restrict__ h, const float* __restrict__ zr,
                            __bf16* __restrict__ xcat, __bf16* __restrict__ xg,
                            int Cx, int Cpad, int KCp, int t, int stage) {
    const int total = BB * NN * Cpad;
    for (int idx = blockIdx.x * blockDim.x + threadIdx.x; idx < total;
         idx += gridDim.x * blockDim.x) {
        const int c = idx % Cpad;
        const int n = (idx / Cpad) % NN;
        const int b = idx / (Cpad * NN);
        float v = 0.f;
        if (c < Cx) {
            const size_t si = (((size_t)b * TT + t) * NN + n) * Cx + c;
            v = xf ? xf[si] : (float)ybf[si];
        } else if (c < Cx + HH) {
            const int o = c - Cx;
            float hv = h[((size_t)b * NN + n) * HH + o];
            if (stage) hv *= zr[((size_t)b * NN + n) * (2 * HH) + o];   // z * h
            v = hv;
        }
        const __bf16 bv = (__bf16)v;
        xcat[((size_t)b * NN + n) * Cpad + c] = bv;
        xg[((size_t)b * NN + n) * KCp + c]    = bv;                     // k = 0 slot
    }
}

// ---------------------------------------------------------------------------
// Graph conv (the hot GEMM), both supports fused per wave:
//   xg[b][n][1][:] = A  @ xcat[b]
//   xg[b][n][2][:] = T2 @ xcat[b]
// One wave per (16-node tile, CT*16 channels, b).  Every B-fragment (xcatT)
// feeds two WMMAs (one per support); A-fragments reused across CT c-tiles.
// ---------------------------------------------------------------------------
template <int CT, int KCp>
__global__ void graph_wmma(const __bf16* __restrict__ A1, const __bf16* __restrict__ A2,
                           const __bf16* __restrict__ xT, __bf16* __restrict__ xg,
                           int Cpad) {
    const int lane = threadIdx.x;                 // 32 threads, full EXEC
    const int mt = blockIdx.x;
    const int cg = blockIdx.y;
    const int b  = blockIdx.z;
    const __bf16* xTb = xT + (size_t)b * Cpad * NN;
    const int row  = mt * 16 + (lane & 15);
    const int aoff = (lane >> 4) * 8;
    const int boff = (lane >> 4) * 8;

    v8f acc[2][CT];
    #pragma unroll
    for (int s = 0; s < 2; ++s)
        #pragma unroll
        for (int i = 0; i < CT; ++i) acc[s][i] = (v8f){0.f,0.f,0.f,0.f,0.f,0.f,0.f,0.f};

    for (int kk = 0; kk < NN; kk += 32) {
        const __bf16* a1p = A1 + (size_t)row * NN + kk + aoff;
        const __bf16* a2p = A2 + (size_t)row * NN + kk + aoff;
        const v16bf af1 = cat16(*(const v8bf*)a1p, *(const v8bf*)(a1p + 16));
        const v16bf af2 = cat16(*(const v8bf*)a2p, *(const v8bf*)(a2p + 16));
        #pragma unroll
        for (int ct = 0; ct < CT; ++ct) {
            const int c = cg * (CT * 16) + ct * 16 + (lane & 15);
            const __bf16* bp = xTb + (size_t)c * NN + kk + boff;
            const v16bf bf = cat16(*(const v8bf*)bp, *(const v8bf*)(bp + 16));
            acc[0][ct] = wmma_bf16(af1, bf, acc[0][ct]);
            acc[1][ct] = wmma_bf16(af2, bf, acc[1][ct]);
        }
    }
    #pragma unroll
    for (int s = 0; s < 2; ++s)
        #pragma unroll
        for (int ct = 0; ct < CT; ++ct)
            #pragma unroll
            for (int i = 0; i < 8; ++i) {
                const int m = mt * 16 + i + ((lane >> 4) ? 8 : 0);
                const int c = cg * (CT * 16) + ct * 16 + (lane & 15);
                xg[((size_t)b * NN + m) * KCp + (s + 1) * Cpad + c] = (__bf16)acc[s][ct][i];
            }
}

// ---------------------------------------------------------------------------
// Per-node contraction + activation (+ fused GRU update in cand mode).
//   Out[b, o] = xg[b][n][0:KC] . Wt[n][o][0:KC] + bias[n][o]
// mode 0: zr = sigmoid(out)            (O = 2H)
// mode 1: c = tanh(out); h = r*h+(1-r)*c; emit sequence output   (O = H)
// One wave per (node, 4 output tiles); M = 32 batch rows = 2 M-tiles.
// ---------------------------------------------------------------------------
__global__ void node_wmma(const __bf16* __restrict__ xg, const __bf16* __restrict__ Wt,
                          const float* __restrict__ bias, float* __restrict__ zr,
                          float* __restrict__ h, __bf16* __restrict__ ybf,
                          float* __restrict__ yf, int KCp, int O, int mode, int t) {
    constexpr int OT = 4;
    const int lane = threadIdx.x;
    const int n  = blockIdx.x;
    const int og = blockIdx.y;
    const int aoff = (lane >> 4) * 8;
    const int boff = (lane >> 4) * 8;

    v8f acc[2][OT];
    #pragma unroll
    for (int m = 0; m < 2; ++m)
        #pragma unroll
        for (int i = 0; i < OT; ++i) acc[m][i] = (v8f){0.f,0.f,0.f,0.f,0.f,0.f,0.f,0.f};

    for (int kc = 0; kc < KCp; kc += 32) {
        v16bf af[2];
        #pragma unroll
        for (int m = 0; m < 2; ++m) {
            const int brow = m * 16 + (lane & 15);
            const __bf16* ap = xg + ((size_t)brow * NN + n) * KCp + kc + aoff;
            af[m] = cat16(*(const v8bf*)ap, *(const v8bf*)(ap + 16));
        }
        #pragma unroll
        for (int ot = 0; ot < OT; ++ot) {
            const int o = og * (OT * 16) + ot * 16 + (lane & 15);
            const __bf16* bp = Wt + ((size_t)n * O + o) * KCp + kc + boff;
            const v16bf bf = cat16(*(const v8bf*)bp, *(const v8bf*)(bp + 16));
            acc[0][ot] = wmma_bf16(af[0], bf, acc[0][ot]);
            acc[1][ot] = wmma_bf16(af[1], bf, acc[1][ot]);
        }
    }

    #pragma unroll
    for (int m = 0; m < 2; ++m)
        #pragma unroll
        for (int ot = 0; ot < OT; ++ot)
            #pragma unroll
            for (int i = 0; i < 8; ++i) {
                const int b = m * 16 + i + ((lane >> 4) ? 8 : 0);
                const int o = og * (OT * 16) + ot * 16 + (lane & 15);
                const float v = acc[m][ot][i] + bias[(size_t)n * O + o];
                if (mode == 0) {
                    zr[((size_t)b * NN + n) * (2 * HH) + o] = sigmoidf_(v);
                } else {
                    const float c  = tanhf(v);
                    const float r  = zr[((size_t)b * NN + n) * (2 * HH) + HH + o];
                    const size_t hi = ((size_t)b * NN + n) * HH + o;
                    const float hn = r * h[hi] + (1.f - r) * c;
                    h[hi] = hn;
                    const size_t yi = (((size_t)b * TT + t) * NN + n) * HH + o;
                    if (ybf) ybf[yi] = (__bf16)hn;
                    else     yf[yi]  = hn;
                }
            }
}

// ---------------------------------------------------------------------------
// Host orchestration
// ---------------------------------------------------------------------------
extern "C" void kernel_launch(void* const* d_in, const int* in_sizes, int n_in,
                              void* d_out, int out_size, void* d_ws, size_t ws_size,
                              hipStream_t stream) {
    (void)in_sizes; (void)n_in; (void)out_size; (void)ws_size;
    const float* x    = (const float*)d_in[0];
    const float* h0in = (const float*)d_in[1];
    const float* Emb  = (const float*)d_in[2];
    const float* wg0  = (const float*)d_in[3];
    const float* bg0  = (const float*)d_in[4];
    const float* wu0  = (const float*)d_in[5];
    const float* bu0  = (const float*)d_in[6];
    const float* wg1  = (const float*)d_in[7];
    const float* bg1  = (const float*)d_in[8];
    const float* wu1  = (const float*)d_in[9];
    const float* bu1  = (const float*)d_in[10];
    float* out = (float*)d_out;

    // ---- workspace carve-out ----
    char* ws = (char*)d_ws;
    size_t off = 0;
    auto carve = [&](size_t bytes) { void* p = ws + off; off += (bytes + 255) & ~size_t(255); return p; };
    __bf16* A_bf   = (__bf16*)carve((size_t)NN * NN * 2);
    __bf16* AT_bf  = (__bf16*)carve((size_t)NN * NN * 2);
    __bf16* T2_bf  = (__bf16*)carve((size_t)NN * NN * 2);
    __bf16* Wg0t   = (__bf16*)carve((size_t)NN * 128 * KC0 * 2);
    __bf16* Wu0t   = (__bf16*)carve((size_t)NN * 64  * KC0 * 2);
    __bf16* Wg1t   = (__bf16*)carve((size_t)NN * 128 * KC1 * 2);
    __bf16* Wu1t   = (__bf16*)carve((size_t)NN * 64  * KC1 * 2);
    float*  Bg0n   = (float*)carve((size_t)NN * 128 * 4);
    float*  Bu0n   = (float*)carve((size_t)NN * 64  * 4);
    float*  Bg1n   = (float*)carve((size_t)NN * 128 * 4);
    float*  Bu1n   = (float*)carve((size_t)NN * 64  * 4);
    float*  hst0   = (float*)carve((size_t)BB * NN * HH * 4);
    float*  hst1   = (float*)carve((size_t)BB * NN * HH * 4);
    __bf16* xcat   = (__bf16*)carve((size_t)BB * NN * CPAD1 * 2);
    __bf16* xcatT  = (__bf16*)carve((size_t)BB * NN * CPAD1 * 2);
    __bf16* xgbuf  = (__bf16*)carve((size_t)BB * NN * KC1 * 2);
    float*  zrbuf  = (float*)carve((size_t)BB * NN * 2 * HH * 4);
    __bf16* y0     = (__bf16*)carve((size_t)BB * TT * NN * HH * 2);

    // ---- one-time setup ----
    adjacency_kernel<<<NN, 256, 0, stream>>>(Emb, A_bf);
    transpose_bf16<<<dim3(NN / 32, NN / 32, 1), dim3(32, 8), 0, stream>>>(A_bf, AT_bf, NN, NN);
    cheb2_kernel<<<dim3(NN / 16, NN / 64), 32, 0, stream>>>(A_bf, AT_bf, T2_bf);

    {
        int tot;
        tot = NN * 128 * KC0;
        mk_weights<<<(tot + 255) / 256, 256, 0, stream>>>(Emb, wg0, Wg0t, DIN + HH, CPAD0, 128, tot);
        tot = NN * 64 * KC0;
        mk_weights<<<(tot + 255) / 256, 256, 0, stream>>>(Emb, wu0, Wu0t, DIN + HH, CPAD0, 64, tot);
        tot = NN * 128 * KC1;
        mk_weights<<<(tot + 255) / 256, 256, 0, stream>>>(Emb, wg1, Wg1t, 2 * HH, CPAD1, 128, tot);
        tot = NN * 64 * KC1;
        mk_weights<<<(tot + 255) / 256, 256, 0, stream>>>(Emb, wu1, Wu1t, 2 * HH, CPAD1, 64, tot);
        mk_bias<<<(NN * 128 + 255) / 256, 256, 0, stream>>>(Emb, bg0, Bg0n, 128);
        mk_bias<<<(NN * 64  + 255) / 256, 256, 0, stream>>>(Emb, bu0, Bu0n, 64);
        mk_bias<<<(NN * 128 + 255) / 256, 256, 0, stream>>>(Emb, bg1, Bg1n, 128);
        mk_bias<<<(NN * 64  + 255) / 256, 256, 0, stream>>>(Emb, bu1, Bu1n, 64);
    }

    const size_t hbytes = (size_t)BB * NN * HH * sizeof(float);
    hipMemcpyAsync(hst0, h0in,                        hbytes, hipMemcpyDeviceToDevice, stream);
    hipMemcpyAsync(hst1, h0in + (size_t)BB * NN * HH, hbytes, hipMemcpyDeviceToDevice, stream);

    // ---- recurrent scan ----
    for (int t = 0; t < TT; ++t) {
        for (int layer = 0; layer < 2; ++layer) {
            const int Cx   = layer ? HH : DIN;
            const int Cpad = layer ? CPAD1 : CPAD0;
            const int KCp  = layer ? KC1 : KC0;
            const __bf16* Wg = layer ? Wg1t : Wg0t;
            const __bf16* Wu = layer ? Wu1t : Wu0t;
            const float*  Bg = layer ? Bg1n : Bg0n;
            const float*  Bu = layer ? Bu1n : Bu0n;
            float* hst = layer ? hst1 : hst0;
            const float*  xf  = layer ? nullptr : x;
            const __bf16* ysr = layer ? y0 : nullptr;

            const int packTot = BB * NN * Cpad;
            const dim3 tgrid(Cpad / 32, NN / 32, BB);
            const dim3 ggrid(NN / 16, 2, BB);   // Cpad/(CT*16) == 2 for both layers

            // --- gate: zr = sigmoid(gconv([x|h])) ---
            pack_kernel<<<(packTot + 255) / 256, 256, 0, stream>>>(
                xf, ysr, hst, zrbuf, xcat, xgbuf, Cx, Cpad, KCp, t, /*stage=*/0);
            transpose_bf16<<<tgrid, dim3(32, 8), 0, stream>>>(xcat, xcatT, NN, Cpad);
            if (layer)
                graph_wmma<4, KC1><<<ggrid, 32, 0, stream>>>(A_bf, T2_bf, xcatT, xgbuf, Cpad);
            else
                graph_wmma<3, KC0><<<ggrid, 32, 0, stream>>>(A_bf, T2_bf, xcatT, xgbuf, Cpad);
            node_wmma<<<dim3(NN, 2), 32, 0, stream>>>(
                xgbuf, Wg, Bg, zrbuf, hst, nullptr, nullptr, KCp, 128, /*mode=*/0, t);

            // --- candidate + GRU update: h = r*h + (1-r)*tanh(gconv([x|z*h])) ---
            pack_kernel<<<(packTot + 255) / 256, 256, 0, stream>>>(
                xf, ysr, hst, zrbuf, xcat, xgbuf, Cx, Cpad, KCp, t, /*stage=*/1);
            transpose_bf16<<<tgrid, dim3(32, 8), 0, stream>>>(xcat, xcatT, NN, Cpad);
            if (layer)
                graph_wmma<4, KC1><<<ggrid, 32, 0, stream>>>(A_bf, T2_bf, xcatT, xgbuf, Cpad);
            else
                graph_wmma<3, KC0><<<ggrid, 32, 0, stream>>>(A_bf, T2_bf, xcatT, xgbuf, Cpad);
            node_wmma<<<dim3(NN, 1), 32, 0, stream>>>(
                xgbuf, Wu, Bu, zrbuf, hst,
                layer ? nullptr : y0, layer ? out : nullptr,
                KCp, 64, /*mode=*/1, t);
        }
    }

    // ---- final hidden states: d_out tail is [2, B, N, H] ----
    float* tail = out + (size_t)BB * TT * NN * HH;
    hipMemcpyAsync(tail,                        hst0, hbytes, hipMemcpyDeviceToDevice, stream);
    hipMemcpyAsync(tail + (size_t)BB * NN * HH, hst1, hbytes, hipMemcpyDeviceToDevice, stream);
}
